// Module2Network_37520834298058
// MI455X (gfx1250) — matrix operand order repaired and checked by reading, not compiled, not run on hardware
//
#include <hip/hip_runtime.h>
#include <math.h>

typedef __attribute__((ext_vector_type(16))) _Float16 v16h;
typedef __attribute__((ext_vector_type(8)))  _Float16 v8h;
typedef __attribute__((ext_vector_type(8)))  float    v8f;

#define NUM_FREQS     64
#define HEAD_DIM      128
#define NUM_BINS      128
#define KAUG          192     // 128 dot dims + 64 magnitude dims
#define ROWS_PER_BLK  128
#define THREADS       256
#define LDS_STRIDE_H  200     // halves per augmented row (400 B, padded for bank-conflict-free b128)
#define EPSF          1e-8f

// ---------------------------------------------------------------------------
// Prep: build fused weight matrix W^T (NUM_BINS x KAUG) in f16.
//   W^T[bin, f]      = rotated_re = (re*cos - im*sin) * inv_norm
//   W^T[bin, 64+f]   = rotated_im = (im*cos + re*sin) * inv_norm
//   W^T[bin, 128+f]  = u_weights[bin, f]
// ---------------------------------------------------------------------------
__global__ void build_w_kernel(const float* __restrict__ angles,
                               const float* __restrict__ probes,
                               const float* __restrict__ u_w,
                               _Float16* __restrict__ Wt) {
    __shared__ float red[64];
    const int bin = blockIdx.x;
    const int f   = threadIdx.x;            // 0..63

    float re = probes[bin * HEAD_DIM + f];
    float im = probes[bin * HEAD_DIM + NUM_FREQS + f];

    red[f] = re * re + im * im;
    __syncthreads();
    #pragma unroll
    for (int s = 32; s > 0; s >>= 1) {
        if (f < s) red[f] += red[f + s];
        __syncthreads();
    }
    const float inv = 1.0f / (sqrtf(red[0]) + EPSF);

    const float th = angles[f];
    const float c  = cosf(th);
    const float s  = sinf(th);

    _Float16* row = Wt + (size_t)bin * KAUG;
    row[f]             = (_Float16)((re * c - im * s) * inv);
    row[NUM_FREQS + f] = (_Float16)((im * c + re * s) * inv);
    row[HEAD_DIM  + f] = (_Float16)(u_w[bin * NUM_FREQS + f]);
}

// ---------------------------------------------------------------------------
// Main: logits = [K, |K|] @ W + bias using V_WMMA_F32_16X16X32_F16.
// One block = 256 threads = 8 wave32, 128 key rows. Each wave owns a 16x128
// output slab: 6 k-tiles x 8 n-tiles = 48 WMMAs.
// ---------------------------------------------------------------------------
__global__ __launch_bounds__(THREADS)
void logits_kernel(const float* __restrict__ Kg,
                   const _Float16* __restrict__ Wt,
                   const float* __restrict__ bias,
                   float* __restrict__ out,
                   int num_keys) {
    __shared__ __align__(16) _Float16 sA[ROWS_PER_BLK * LDS_STRIDE_H];  // 51200 B

    const int tid  = threadIdx.x;
    const int base = blockIdx.x * ROWS_PER_BLK;

    // ---- cooperative load: K row -> f16 + magnitude feature, into LDS ----
    // 128 rows x 64 freq pairs = 8192 items, 32 per thread. Each K element
    // is read from HBM exactly once.
    #pragma unroll 4
    for (int j = 0; j < 32; ++j) {
        const int idx = tid + THREADS * j;
        const int r   = idx >> 6;                 // local row 0..127
        const int f   = idx & 63;                 // freq 0..63
        int rg = base + r;
        if (rg >= num_keys) rg = num_keys - 1;    // clamp: keep EXEC full
        const float re = Kg[(size_t)rg * HEAD_DIM + f];
        const float im = Kg[(size_t)rg * HEAD_DIM + NUM_FREQS + f];
        _Float16* row = sA + r * LDS_STRIDE_H;
        row[f]             = (_Float16)re;
        row[NUM_FREQS + f] = (_Float16)im;
        row[HEAD_DIM  + f] = (_Float16)sqrtf(re * re + im * im + EPSF);
    }
    __syncthreads();

    const int wave = tid >> 5;        // 0..7
    const int lane = tid & 31;
    const int lrow = lane & 15;       // M (A/D) or N (B) within tile
    const int hi   = lane >> 4;       // lane-half selector

    v8f acc[8];
    #pragma unroll
    for (int n = 0; n < 8; ++n)
        acc[n] = (v8f){0.f, 0.f, 0.f, 0.f, 0.f, 0.f, 0.f, 0.f};

    // A fragment base: row = wave*16 + (lane&15); lane-half K offset = hi*8
    const _Float16* aBase = sA + (wave * 16 + lrow) * LDS_STRIDE_H + hi * 8;

    #pragma unroll
    for (int kt = 0; kt < 6; ++kt) {
        // 16-bit A 16x32 layout: lanes<16 hold K {kb..kb+7, kb+16..kb+23},
        // lanes>=16 hold K {kb+8..kb+15, kb+24..kb+31}; 2 halves per VGPR.
        const _Float16* ap = aBase + kt * 32;
        const v8h a0 = *(const v8h*)(ap);
        const v8h a1 = *(const v8h*)(ap + 16);
        const v16h a = __builtin_shufflevector(a0, a1,
            0, 1, 2, 3, 4, 5, 6, 7, 8, 9, 10, 11, 12, 13, 14, 15);

        #pragma unroll
        for (int nt = 0; nt < 8; ++nt) {
            // B 32x16 layout: lane holds column N=lane&15, K {kb..kb+7, kb+16..kb+23}
            // (lanes<16) or {kb+8..kb+15, kb+24..kb+31} (lanes>=16), matching A.
            const _Float16* bp = Wt + (size_t)(nt * 16 + lrow) * KAUG
                                    + kt * 32 + hi * 8;
            const v8h b0 = *(const v8h*)(bp);
            const v8h b1 = *(const v8h*)(bp + 16);
            const v16h b = __builtin_shufflevector(b0, b1,
                0, 1, 2, 3, 4, 5, 6, 7, 8, 9, 10, 11, 12, 13, 14, 15);

            acc[nt] = __builtin_amdgcn_wmma_f32_16x16x32_f16(
                false, a, false, b, (short)0, acc[nt], false, false);
        }
    }

    // ---- store with bias; C/D layout: VGPR v -> M = v + 8*hi, N = lane&15 ----
    const int mrow0 = base + wave * 16 + hi * 8;
    #pragma unroll
    for (int nt = 0; nt < 8; ++nt) {
        const int   ncol = nt * 16 + lrow;
        const float bv   = bias[ncol];
        #pragma unroll
        for (int v = 0; v < 8; ++v) {
            const int m = mrow0 + v;
            if (m < num_keys)
                out[(size_t)m * NUM_BINS + ncol] = acc[nt][v] + bv;
        }
    }
}

extern "C" void kernel_launch(void* const* d_in, const int* in_sizes, int n_in,
                              void* d_out, int out_size, void* d_ws, size_t ws_size,
                              hipStream_t stream) {
    const float* Kg     = (const float*)d_in[0];   // (num_keys, 128)
    const float* angles = (const float*)d_in[1];   // (64,)
    const float* probes = (const float*)d_in[2];   // (128, 128)
    const float* u_w    = (const float*)d_in[3];   // (128, 64)
    const float* k_bias = (const float*)d_in[4];   // (128,)
    float*       out    = (float*)d_out;

    const int num_keys = in_sizes[0] / HEAD_DIM;

    _Float16* Wt = (_Float16*)d_ws;                // NUM_BINS*KAUG*2 = 49152 B

    build_w_kernel<<<NUM_BINS, 64, 0, stream>>>(angles, probes, u_w, Wt);

    const int blocks = (num_keys + ROWS_PER_BLK - 1) / ROWS_PER_BLK;
    logits_kernel<<<blocks, THREADS, 0, stream>>>(Kg, Wt, k_bias, out, num_keys);
}
